// ScaledDotProductAttention_79104707658327
// MI455X (gfx1250) — compile-verified, matrix operand order checked by reading, not run on hardware
//
#include <hip/hip_runtime.h>
#include <hip/hip_bf16.h>

typedef __bf16 v16bf __attribute__((ext_vector_type(16)));
typedef __bf16 v8bf  __attribute__((ext_vector_type(8)));
typedef float  v8f   __attribute__((ext_vector_type(8)));

#define BATCH 16
#define SEQ   4096
#define DIM   64
#define WAVES 4
#define SCALE2 0.18033688011112042f   // (1/sqrt(64)) * log2(e): softmax in log2 domain
#define NEGINF (-3.0e38f)

// Raw v_exp_f32 (computes 2^x) — avoids libm name issues entirely.
__device__ __forceinline__ float exp2_fast(float x) {
  return __builtin_amdgcn_exp2f(x);
}

__device__ __forceinline__ v8f wmma_bf16(v16bf a, v16bf b, v8f c) {
  return __builtin_amdgcn_wmma_f32_16x16x32_bf16(false, a, false, b,
                                                 (short)0, c, false, false);
}

// f32 source: two contiguous 8-float runs -> 4x global_load_b128 + packed cvt.
// ISA 7.12.2 16-bit A/B layout: lane<16 -> K = kofs+[0..7], kofs+16+[0..7];
// lane>=16 -> runs shifted by kbase=8.
__device__ __forceinline__ v16bf load_row_frag(const float* __restrict__ rowptr,
                                               int kofs, int kbase) {
  const float* a = rowptr + kofs + kbase;
  const float* b = rowptr + kofs + 16 + kbase;
  float4 a0 = *(const float4*)(a);
  float4 a1 = *(const float4*)(a + 4);
  float4 b0 = *(const float4*)(b);
  float4 b1 = *(const float4*)(b + 4);
  v16bf f;
  f[0]  = (__bf16)a0.x; f[1]  = (__bf16)a0.y; f[2]  = (__bf16)a0.z; f[3]  = (__bf16)a0.w;
  f[4]  = (__bf16)a1.x; f[5]  = (__bf16)a1.y; f[6]  = (__bf16)a1.z; f[7]  = (__bf16)a1.w;
  f[8]  = (__bf16)b0.x; f[9]  = (__bf16)b0.y; f[10] = (__bf16)b0.z; f[11] = (__bf16)b0.w;
  f[12] = (__bf16)b1.x; f[13] = (__bf16)b1.y; f[14] = (__bf16)b1.z; f[15] = (__bf16)b1.w;
  return f;
}

// Packed bf16 source: the same two runs are 16B each -> exactly 2x
// global_load_b128, zero conversions.
__device__ __forceinline__ v16bf load_row_frag(const __bf16* __restrict__ rowptr,
                                               int kofs, int kbase) {
  v8bf a = *(const v8bf*)(rowptr + kofs + kbase);
  v8bf b = *(const v8bf*)(rowptr + kofs + 16 + kbase);
  return __builtin_shufflevector(a, b, 0, 1, 2, 3, 4, 5, 6, 7,
                                       8, 9, 10, 11, 12, 13, 14, 15);
}

// Fallback (f32, untransposed V): strided b32 loads, each fetching full 64B lines.
__device__ __forceinline__ v16bf load_v_frag_strided(const float* __restrict__ vbase,
                                                     int tbase, int kbase, int col) {
  v16bf f;
  const float* p0 = vbase + (size_t)(tbase + kbase) * DIM + col;
  const float* p1 = vbase + (size_t)(tbase + 16 + kbase) * DIM + col;
#pragma unroll
  for (int i = 0; i < 8; ++i) f[i]     = (__bf16)p0[(size_t)i * DIM];
#pragma unroll
  for (int i = 0; i < 8; ++i) f[8 + i] = (__bf16)p1[(size_t)i * DIM];
  return f;
}

// ---------------- pack kernels (one cheap 24MB pass vs 1GB attn traffic) ----------------
__global__ void pack_rowmajor(const float* __restrict__ src, __bf16* __restrict__ dst,
                              int n8) {
  const int i = blockIdx.x * blockDim.x + threadIdx.x;
  if (i >= n8) return;
  const float4* s = (const float4*)src + (size_t)i * 2;
  float4 x = s[0], y = s[1];
  v8bf o;
  o[0] = (__bf16)x.x; o[1] = (__bf16)x.y; o[2] = (__bf16)x.z; o[3] = (__bf16)x.w;
  o[4] = (__bf16)y.x; o[5] = (__bf16)y.y; o[6] = (__bf16)y.z; o[7] = (__bf16)y.w;
  *(v8bf*)(dst + (size_t)i * 8) = o;
}

__global__ void pack_vt(const float* __restrict__ v, __bf16* __restrict__ vt) {
  const int idx = blockIdx.x * blockDim.x + threadIdx.x;   // BATCH*DIM*(SEQ/8)
  if (idx >= BATCH * DIM * (SEQ / 8)) return;
  const int t8 = idx & (SEQ / 8 - 1);
  const int d  = (idx >> 9) & (DIM - 1);
  const int b  = idx >> 15;
  const float* src = v + ((size_t)b * SEQ + (size_t)t8 * 8) * DIM + d;
  v8bf o;
#pragma unroll
  for (int i = 0; i < 8; ++i) o[i] = (__bf16)src[(size_t)i * DIM];
  *(v8bf*)(vt + ((size_t)b * DIM + d) * SEQ + (size_t)t8 * 8) = o;
}

// ---------------- main kernel ----------------
// T = __bf16: q,k packed bf16 row-major; vsrc = bf16 V^T [b][d][t].
// T = float : original layouts, inline conversion (workspace fallback).
template <typename T>
__global__ void __launch_bounds__(WAVES * 32)
causal_attn_kernel(const T* __restrict__ q, const T* __restrict__ k,
                   const T* __restrict__ vsrc, float* __restrict__ out,
                   float* __restrict__ attn) {
  constexpr bool PACKED = (sizeof(T) == 2);

  const int lane   = threadIdx.x & 31;
  const int wave   = threadIdx.x >> 5;
  const int lane16 = lane & 15;
  const int half   = lane >> 4;
  const int kbase  = half * 8;

  const int tile  = blockIdx.x * WAVES + wave;
  const int b     = tile >> 8;
  const int rtile = tile & 255;
  const int rbase = rtile * 16;

  const T* qrow = q + ((size_t)b * SEQ + rbase + lane16) * DIM;
  const T* kbp  = k + (size_t)b * SEQ * DIM;
  const T* vbp  = vsrc + (size_t)b * SEQ * DIM;   // V or V^T (same extent)
  float* outb   = out + (size_t)b * SEQ * DIM;
  float* attnb  = attn + (size_t)b * SEQ * SEQ;

  const v8f zero8 = {};
  // Q as B operand of S^T = K_tile x Q^T (lane <-> query row m).
  const v16bf bQ0 = load_row_frag(qrow, 0, kbase);
  const v16bf bQ1 = load_row_frag(qrow, 32, kbase);

  const int r    = rbase + lane16;
  const int rloc = r - 8 * half;      // score reg j covers ct = tbase + j + 8*half

  float mrun = NEGINF, lrun = 0.0f;

  // ---- Pass 1: unmasked tiles (mask only matters on the diagonal) ----
  for (int jt = 0; jt < rtile; ++jt) {
    const int tbase = jt * 16;
    __builtin_prefetch(kbp + (size_t)(tbase + 16 + lane16) * DIM, 0, 0);
    const T* krow = kbp + (size_t)(tbase + lane16) * DIM;
    v8f c = wmma_bf16(load_row_frag(krow, 0, kbase), bQ0, zero8);
    c = wmma_bf16(load_row_frag(krow, 32, kbase), bQ1, c);
    float s[8];
#pragma unroll
    for (int j = 0; j < 8; ++j) s[j] = c[j] * SCALE2;
    float tmax = s[0];
#pragma unroll
    for (int j = 1; j < 8; ++j) tmax = fmaxf(tmax, s[j]);
    const float mnew = fmaxf(mrun, tmax);
    float esum = 0.0f;
#pragma unroll
    for (int j = 0; j < 8; ++j) esum += exp2_fast(s[j] - mnew);
    lrun = lrun * exp2_fast(mrun - mnew) + esum;
    mrun = mnew;
  }
  {  // diagonal tile (the only masked one)
    const int tbase = rbase;
    const T* krow = kbp + (size_t)(tbase + lane16) * DIM;
    v8f c = wmma_bf16(load_row_frag(krow, 0, kbase), bQ0, zero8);
    c = wmma_bf16(load_row_frag(krow, 32, kbase), bQ1, c);
    float s[8];
#pragma unroll
    for (int j = 0; j < 8; ++j)
      s[j] = (tbase + j <= rloc) ? c[j] * SCALE2 : NEGINF;
    float tmax = s[0];
#pragma unroll
    for (int j = 1; j < 8; ++j) tmax = fmaxf(tmax, s[j]);
    const float mnew = fmaxf(mrun, tmax);
    float esum = 0.0f;
#pragma unroll
    for (int j = 0; j < 8; ++j) esum += exp2_fast(s[j] - mnew);
    lrun = lrun * exp2_fast(mrun - mnew) + esum;
    mrun = mnew;
  }

  // Merge half-stats: the only cross-lane ops of the whole softmax.
  const float mo   = __shfl_xor(mrun, 16, 32);
  const float lo   = __shfl_xor(lrun, 16, 32);
  const float mfin = fmaxf(mrun, mo);
  const float invl = 1.0f / (lrun * exp2_fast(mrun - mfin) + lo * exp2_fast(mo - mfin));
  const float negm = -mfin;

  // ---- Pass 2: single write of attn + out = P @ V ----
  v8f cO[4] = {zero8, zero8, zero8, zero8};

  auto do_pair = [&](int colbase, bool masked) {
    v16bf aP;   // transposed C layout == A-fragment layout for P (16m x 32t)
#pragma unroll
    for (int sub = 0; sub < 2; ++sub) {
      const int tbase = colbase + sub * 16;
      const T* krow = kbp + (size_t)(tbase + lane16) * DIM;
      v8f c = wmma_bf16(load_row_frag(krow, 0, kbase), bQ0, zero8);
      c = wmma_bf16(load_row_frag(krow, 32, kbase), bQ1, c);
      float p[8];
#pragma unroll
      for (int j = 0; j < 8; ++j) {
        const float e = exp2_fast(__builtin_fmaf(c[j], SCALE2, negm)) * invl;
        p[j] = (masked && (tbase + j > rloc)) ? 0.0f : e;
        aP[sub * 8 + j] = (__bf16)p[j];
      }
      float* rowp = attnb + (size_t)r * SEQ + tbase + 8 * half;
      *(float4*)(rowp)     = make_float4(p[0], p[1], p[2], p[3]);
      *(float4*)(rowp + 4) = make_float4(p[4], p[5], p[6], p[7]);
    }
#pragma unroll
    for (int ch = 0; ch < 4; ++ch) {
      v16bf bV;
      if constexpr (PACKED) {
        // V^T row d: slots t = colbase + kbase + [0..7], + 16 -> 2x b128
        bV = load_row_frag(vbp + (size_t)(ch * 16 + lane16) * SEQ, colbase, kbase);
      } else {
        bV = load_v_frag_strided(vbp, colbase, kbase, ch * 16 + lane16);
      }
      cO[ch] = wmma_bf16(aP, bV, cO[ch]);
    }
  };

  const int dp = rtile >> 1;                       // diagonal 32-col pair
  for (int pair = 0; pair < dp; ++pair) do_pair(pair * 32, false);
  do_pair(dp * 32, true);

  // Fully masked pairs: probs are 0 and d_out is poisoned -> store zeros.
  const float4 z = make_float4(0.0f, 0.0f, 0.0f, 0.0f);
  for (int pair = dp + 1; pair < SEQ / 32; ++pair) {
    float* rowp = attnb + (size_t)r * SEQ + pair * 32 + 8 * half;
    *(float4*)(rowp)      = z;
    *(float4*)(rowp + 4)  = z;
    *(float4*)(rowp + 16) = z;
    *(float4*)(rowp + 20) = z;
  }

  // out tile: C/D layout (reg j <-> row m, lane <-> d) -> row-major.
#pragma unroll
  for (int ch = 0; ch < 4; ++ch) {
#pragma unroll
    for (int j = 0; j < 8; ++j) {
      const int rr = rbase + j + 8 * half;
      outb[(size_t)rr * DIM + ch * 16 + lane16] = cO[ch][j];
    }
  }
}

extern "C" void kernel_launch(void* const* d_in, const int* in_sizes, int n_in,
                              void* d_out, int out_size, void* d_ws, size_t ws_size,
                              hipStream_t stream) {
  const float* q = (const float*)d_in[0];
  const float* k = (const float*)d_in[1];
  const float* v = (const float*)d_in[2];
  float* out  = (float*)d_out;
  float* attn = (float*)d_out + (size_t)BATCH * SEQ * DIM;

  dim3 grid(BATCH * (SEQ / 16) / WAVES);   // 1024 blocks
  dim3 block(WAVES * 32);                  // 4 waves (wave32)

  const size_t nelem = (size_t)BATCH * SEQ * DIM;          // 4M per tensor
  if (ws_size >= nelem * 2 * 3) {                          // 24 MB: packed path
    __bf16* qb = (__bf16*)d_ws;
    __bf16* kb = qb + nelem;
    __bf16* vt = kb + nelem;
    const int n8 = (int)(nelem / 8);
    pack_rowmajor<<<(n8 + 255) / 256, 256, 0, stream>>>(q, qb, n8);
    pack_rowmajor<<<(n8 + 255) / 256, 256, 0, stream>>>(k, kb, n8);
    pack_vt<<<(BATCH * DIM * (SEQ / 8) + 255) / 256, 256, 0, stream>>>(v, vt);
    causal_attn_kernel<__bf16><<<grid, block, 0, stream>>>(qb, kb, vt, out, attn);
  } else {                                                 // fallback: inline cvt
    causal_attn_kernel<float><<<grid, block, 0, stream>>>(q, k, v, out, attn);
  }
}
